// ShapGATModule_14757507629511
// MI455X (gfx1250) — compile-verified, matrix operand order checked by reading, not run on hardware
//
#include <hip/hip_runtime.h>
#include <hip/hip_bf16.h>

typedef __attribute__((ext_vector_type(16))) _Float16 v16h;
typedef __attribute__((ext_vector_type(8)))  float    v8f;
typedef _Float16 f16;

#define N_NODES 50000
#define N_EDGES 400000
#define N_ETOT  450000   // + self loops
#define N_GRAPH 1000
#define NPG     50
#define D_IN    128
#define HC1     256      // heads*hid = 4*64
#define HEADS   4
#define C1      64
#define K1      160      // 129 padded to 5*32
#define KT1     5
#define NT1     16       // 256/16
#define K2      256
#define KT2     8
#define C2      64
#define NT2     4
#define MT      3125     // 50000/16

// ---- fragment K mapping for 16-bit A/B tiles (ISA 7.12.2) ----
// j = 0..15 halves in v16h; v = j>>1 is VGPR, e = j&1 element; half = lane>>4
__device__ __forceinline__ int kmap(int j, int half) {
    int v = j >> 1, e = j & 1;
    return ((v >> 2) << 4) + (half << 3) + ((v & 3) << 1) + e;
}

// ordered-uint encoding for float atomicMax
__device__ __forceinline__ unsigned fkey(float f) {
    unsigned u = __float_as_uint(f);
    return (u & 0x80000000u) ? ~u : (u | 0x80000000u);
}
__device__ __forceinline__ float funkey(unsigned k) {
    unsigned u = (k & 0x80000000u) ? (k ^ 0x80000000u) : ~k;
    return __uint_as_float(u);
}

// ------------------------------------------------------------------ utils
__global__ void k_zero(float* p, size_t n) {
    size_t i = (size_t)blockIdx.x * blockDim.x + threadIdx.x;
    size_t s = (size_t)gridDim.x * blockDim.x;
    for (; i < n; i += s) p[i] = 0.f;
}

__global__ void k_mean(const float* __restrict__ eattr, float* sums) {
    __shared__ float s0[256], s1[256];
    float a0 = 0.f, a1 = 0.f;
    for (int e = blockIdx.x * blockDim.x + threadIdx.x; e < N_EDGES;
         e += gridDim.x * blockDim.x) {
        a0 += eattr[2 * e];
        a1 += eattr[2 * e + 1];
    }
    s0[threadIdx.x] = a0; s1[threadIdx.x] = a1;
    __syncthreads();
    for (int w = 128; w > 0; w >>= 1) {
        if ((int)threadIdx.x < w) {
            s0[threadIdx.x] += s0[threadIdx.x + w];
            s1[threadIdx.x] += s1[threadIdx.x + w];
        }
        __syncthreads();
    }
    if (threadIdx.x == 0) { atomicAdd(&sums[0], s0[0]); atomicAdd(&sums[1], s1[0]); }
}

// coef layout (at `coef`): [c0[4]][c1[4]][aeloop[4]][d0][d1][aeloop2]
__global__ void k_coef(const float* We1, const float* atte1,
                       const float* We2, const float* atte2,
                       const float* sums, float* coef) {
    __shared__ float c0[HEADS], c1[HEADS], d01[2];
    int t = threadIdx.x;
    if (t < HEADS) {
        float s = 0.f;
        for (int c = 0; c < C1; ++c) s += We1[t * C1 + c] * atte1[t * C1 + c];
        c0[t] = s;
        s = 0.f;
        for (int c = 0; c < C1; ++c) s += We1[HC1 + t * C1 + c] * atte1[t * C1 + c];
        c1[t] = s;
    }
    if (t == HEADS) {
        float s = 0.f;
        for (int c = 0; c < C2; ++c) s += We2[c] * atte2[c];
        d01[0] = s;
        s = 0.f;
        for (int c = 0; c < C2; ++c) s += We2[C2 + c] * atte2[c];
        d01[1] = s;
    }
    __syncthreads();
    if (t == 0) {
        float m0 = sums[0] / (float)N_EDGES, m1 = sums[1] / (float)N_EDGES;
        for (int h = 0; h < HEADS; ++h) {
            coef[h] = c0[h];
            coef[HEADS + h] = c1[h];
            coef[2 * HEADS + h] = m0 * c0[h] + m1 * c1[h];
        }
        coef[12] = d01[0];
        coef[13] = d01[1];
        coef[14] = m0 * d01[0] + m1 * d01[1];
    }
}

// ------------------------------------------------------------------ packing
// A fragments: Apack[((tile*KT + kt)*32 + lane)*16 + j]
__global__ void k_packA1(const float* __restrict__ nf, const int* __restrict__ pm,
                         f16* __restrict__ Apack) {
    int idx = blockIdx.x * blockDim.x + threadIdx.x;
    if (idx >= MT * KT1 * 512) return;
    int j = idx & 15, lane = (idx >> 4) & 31;
    int t = idx >> 9;
    int kt = t % KT1, tile = t / KT1;
    int half = lane >> 4;
    int row = tile * 16 + (lane & 15);
    int k = kt * 32 + kmap(j, half);
    float v = 0.f;
    if (k < D_IN) v = nf[(size_t)row * D_IN + k];
    else if (k == D_IN)
        v = (row == N_NODES - 1 || pm[row + 1] != pm[row]) ? 1.f : 0.f;
    Apack[idx] = (f16)v;
}

__global__ void k_packB(const float* __restrict__ W, f16* __restrict__ Bpack,
                        int Kvalid, int Ncols, int KT, int NT) {
    int idx = blockIdx.x * blockDim.x + threadIdx.x;
    if (idx >= NT * KT * 512) return;
    int j = idx & 15, lane = (idx >> 4) & 31;
    int t = idx >> 9;
    int kt = t % KT, nt = t / KT;
    int half = lane >> 4;
    int col = nt * 16 + (lane & 15);
    int k = kt * 32 + kmap(j, half);
    Bpack[idx] = (f16)((k < Kvalid) ? W[(size_t)k * Ncols + col] : 0.f);
}

// ------------------------------------------------------------------ WMMA GEMM
// one wave per 16x16 output tile; C is [MT*16, Nt*16] row-major (ldc)
__global__ void k_gemm(const f16* __restrict__ Apack, const f16* __restrict__ Bpack,
                       float* __restrict__ C, int Nt, int Kt, int ldc) {
    int wid  = (blockIdx.x * blockDim.x + threadIdx.x) >> 5;
    int lane = threadIdx.x & 31;
    if (wid >= MT * Nt) return;                  // whole-wave uniform exit
    int mtile = wid / Nt, ntile = wid % Nt;
    const f16* ap = Apack + (size_t)mtile * Kt * 512 + (size_t)lane * 16;
    const f16* bp = Bpack + (size_t)ntile * Kt * 512 + (size_t)lane * 16;
    v8f acc = {};
    for (int kt = 0; kt < Kt; ++kt) {
        if (kt + 1 < Kt) {
            __builtin_prefetch(ap + (size_t)(kt + 1) * 512, 0, 1);
            __builtin_prefetch(bp + (size_t)(kt + 1) * 512, 0, 1);
        }
        v16h a = *(const v16h*)(ap + (size_t)kt * 512);
        v16h b = *(const v16h*)(bp + (size_t)kt * 512);
        acc = __builtin_amdgcn_wmma_f32_16x16x32_f16(
            false, a, false, b, (short)0, acc, false, false);
    }
    int col   = ntile * 16 + (lane & 15);
    int rbase = mtile * 16 + ((lane >> 4) << 3);
#pragma unroll
    for (int i = 0; i < 8; ++i)
        C[(size_t)(rbase + i) * ldc + col] = acc[i];
}

// ------------------------------------------------------------------ attention
__global__ void k_attn1(const float* __restrict__ xp, const float* __restrict__ atts,
                        const float* __restrict__ attd, float* asrc, float* adst) {
    int id = blockIdx.x * blockDim.x + threadIdx.x;
    if (id >= N_NODES * HEADS) return;
    int n = id >> 2, h = id & 3;
    const float* xr = xp + (size_t)n * HC1 + h * C1;
    float s = 0.f, d = 0.f;
    for (int c = 0; c < C1; ++c) {
        float v = xr[c];
        s += v * atts[h * C1 + c];
        d += v * attd[h * C1 + c];
    }
    asrc[id] = s; adst[id] = d;
}

__global__ void k_attn2(const float* __restrict__ xp, const float* __restrict__ atts,
                        const float* __restrict__ attd, float* asrc, float* adst) {
    int n = blockIdx.x * blockDim.x + threadIdx.x;
    if (n >= N_NODES) return;
    const float* xr = xp + (size_t)n * C2;
    float s = 0.f, d = 0.f;
    for (int c = 0; c < C2; ++c) {
        float v = xr[c];
        s += v * atts[c];
        d += v * attd[c];
    }
    asrc[n] = s; adst[n] = d;
}

__device__ __forceinline__ void edge_sd(const int* ei, int e, int& src, int& dst) {
    if (e < N_EDGES) { src = ei[e]; dst = ei[N_EDGES + e]; }
    else             { src = dst = e - N_EDGES; }
}

__device__ __forceinline__ float edge_alpha(const int* ei, const float* eattr,
                                            const float* asrc, const float* adst,
                                            const float* coef, int e, int h, int H,
                                            int src, int dst) {
    float ae;
    if (e < N_EDGES) ae = eattr[2 * e] * coef[h] + eattr[2 * e + 1] * coef[H + h];
    else             ae = coef[2 * H + h];
    float a = asrc[src * H + h] + adst[dst * H + h] + ae;
    return a >= 0.f ? a : 0.2f * a;   // LeakyReLU(0.2)
}

__global__ void k_edge_max(const int* __restrict__ ei, const float* __restrict__ eattr,
                           const float* __restrict__ asrc, const float* __restrict__ adst,
                           const float* __restrict__ coef, unsigned* mkey, int H) {
    int id = blockIdx.x * blockDim.x + threadIdx.x;
    if (id >= N_ETOT * H) return;
    int e = id / H, h = id - e * H;
    int src, dst; edge_sd(ei, e, src, dst);
    float a = edge_alpha(ei, eattr, asrc, adst, coef, e, h, H, src, dst);
    atomicMax(&mkey[dst * H + h], fkey(a));
}

__global__ void k_edge_exp(const int* __restrict__ ei, const float* __restrict__ eattr,
                           const float* __restrict__ asrc, const float* __restrict__ adst,
                           const float* __restrict__ coef, const unsigned* __restrict__ mkey,
                           float* __restrict__ ex, float* den, int H) {
    int id = blockIdx.x * blockDim.x + threadIdx.x;
    if (id >= N_ETOT * H) return;
    int e = id / H, h = id - e * H;
    int src, dst; edge_sd(ei, e, src, dst);
    float a = edge_alpha(ei, eattr, asrc, adst, coef, e, h, H, src, dst);
    float m = funkey(mkey[dst * H + h]);
    float v = __expf(a - m);
    ex[id] = v;
    atomicAdd(&den[dst * H + h], v);
}

// conv1 scatter: one 256-thread block per edge (coalesced row read, f32 atomics)
__global__ void k_scatter1(const int* __restrict__ ei, const float* __restrict__ xp,
                           const float* __restrict__ ex, const float* __restrict__ den,
                           float* __restrict__ agg) {
    int e = blockIdx.x, c = threadIdx.x, h = c >> 6;
    int src, dst; edge_sd(ei, e, src, dst);
    float attn = ex[(size_t)e * HEADS + h] / (den[dst * HEADS + h] + 1e-16f);
    atomicAdd(&agg[(size_t)dst * HC1 + c], xp[(size_t)src * HC1 + c] * attn);
}

__global__ void k_scatter2(const int* __restrict__ ei, const float* __restrict__ xp,
                           const float* __restrict__ ex, const float* __restrict__ den,
                           float* __restrict__ agg) {
    size_t id = (size_t)blockIdx.x * blockDim.x + threadIdx.x;
    if (id >= (size_t)N_ETOT * C2) return;
    int e = (int)(id >> 6), c = (int)(id & 63);
    int src, dst; edge_sd(ei, e, src, dst);
    float attn = ex[e] / (den[dst] + 1e-16f);
    atomicAdd(&agg[(size_t)dst * C2 + c], xp[(size_t)src * C2 + c] * attn);
}

// h = relu(agg1 + b1), repacked straight into conv2 A fragments (f16)
__global__ void k_epi1(const float* __restrict__ agg1, const float* __restrict__ b1,
                       f16* __restrict__ Apack2) {
    int idx = blockIdx.x * blockDim.x + threadIdx.x;
    if (idx >= MT * KT2 * 512) return;
    int j = idx & 15, lane = (idx >> 4) & 31;
    int t = idx >> 9;
    int kt = t % KT2, tile = t / KT2;
    int half = lane >> 4;
    int row = tile * 16 + (lane & 15);
    int k = kt * 32 + kmap(j, half);
    float v = agg1[(size_t)row * HC1 + k] + b1[k];
    Apack2[idx] = (f16)(v > 0.f ? v : 0.f);
}

// masked segment-mean pooling; graphs are contiguous blocks of NPG nodes
__global__ void k_pool(const float* __restrict__ agg2, const float* __restrict__ b2,
                       const int* __restrict__ pm, float* __restrict__ out) {
    int g = blockIdx.x, c = threadIdx.x;
    float ms = 0.f, ss = 0.f, mc = 0.f, sc = 0.f;
    for (int i = 0; i < NPG; ++i) {
        int n = g * NPG + i;
        int tag = (n == N_NODES - 1 || pm[n + 1] != pm[n]) ? 1 : 0;
        float v = agg2[(size_t)n * C2 + c] + b2[c];
        if (tag) { ss += v; sc += 1.f; } else { ms += v; mc += 1.f; }
    }
    out[(size_t)g * 128 + c]      = ms / fmaxf(mc, 1.f);
    out[(size_t)g * 128 + 64 + c] = ss / fmaxf(sc, 1.f);
}

// ------------------------------------------------------------------ launch
extern "C" void kernel_launch(void* const* d_in, const int* in_sizes, int n_in,
                              void* d_out, int out_size, void* d_ws, size_t ws_size,
                              hipStream_t stream) {
    (void)in_sizes; (void)n_in; (void)out_size; (void)ws_size;
    const float* nf    = (const float*)d_in[0];
    const int*   ei    = (const int*)d_in[1];
    const float* eattr = (const float*)d_in[2];
    const int*   pm    = (const int*)d_in[3];
    const float* W1    = (const float*)d_in[4];
    const float* We1   = (const float*)d_in[5];
    const float* atts1 = (const float*)d_in[6];
    const float* attd1 = (const float*)d_in[7];
    const float* atte1 = (const float*)d_in[8];
    const float* b1    = (const float*)d_in[9];
    const float* W2    = (const float*)d_in[10];
    const float* We2   = (const float*)d_in[11];
    const float* atts2 = (const float*)d_in[12];
    const float* attd2 = (const float*)d_in[13];
    const float* atte2 = (const float*)d_in[14];
    const float* b2    = (const float*)d_in[15];
    float* out = (float*)d_out;

    char* base = (char*)d_ws;
    size_t o = 0;
    auto alloc = [&](size_t bytes) -> char* {
        char* p = base + o;
        o += (bytes + 255) & ~(size_t)255;
        return p;
    };
    f16*   Apack1 = (f16*)alloc((size_t)MT * KT1 * 512 * 2);
    f16*   Bpack1 = (f16*)alloc((size_t)NT1 * KT1 * 512 * 2);
    f16*   Apack2 = (f16*)alloc((size_t)MT * KT2 * 512 * 2);
    f16*   Bpack2 = (f16*)alloc((size_t)NT2 * KT2 * 512 * 2);
    float* xp1    = (float*)alloc((size_t)N_NODES * HC1 * 4);
    float* xp2    = (float*)alloc((size_t)N_NODES * C2 * 4);
    float* asrc1  = (float*)alloc((size_t)N_NODES * HEADS * 4);
    float* adst1  = (float*)alloc((size_t)N_NODES * HEADS * 4);
    float* asrc2  = (float*)alloc((size_t)N_NODES * 4);
    float* adst2  = (float*)alloc((size_t)N_NODES * 4);
    float* ex1    = (float*)alloc((size_t)N_ETOT * HEADS * 4);
    float* ex2    = (float*)alloc((size_t)N_ETOT * 4);
    // zero-initialized region (contiguous)
    size_t zstart = o;
    float*    agg1 = (float*)alloc((size_t)N_NODES * HC1 * 4);
    float*    agg2 = (float*)alloc((size_t)N_NODES * C2 * 4);
    unsigned* m1   = (unsigned*)alloc((size_t)N_NODES * HEADS * 4);
    float*    den1 = (float*)alloc((size_t)N_NODES * HEADS * 4);
    unsigned* m2   = (unsigned*)alloc((size_t)N_NODES * 4);
    float*    den2 = (float*)alloc((size_t)N_NODES * 4);
    float*    scal = (float*)alloc(64 * 4);   // [0..1]=sums, [2..16]=coefs
    size_t zcount = (o - zstart) / 4;
    float* coef1 = scal + 2;        // c0[4] c1[4] aeloop[4]
    float* coef2 = scal + 2 + 12;   // d0 d1 aeloop2

    // 0) zero accumulators
    k_zero<<<2048, 256, 0, stream>>>((float*)(base + zstart), zcount);
    // 1) edge-attr mean + attention-edge coefficients
    k_mean<<<512, 256, 0, stream>>>(eattr, scal);
    k_coef<<<1, 32, 0, stream>>>(We1, atte1, We2, atte2, scal, coef1);
    // 2) pack fragments
    k_packA1<<<(MT * KT1 * 512 + 255) / 256, 256, 0, stream>>>(nf, pm, Apack1);
    k_packB<<<(NT1 * KT1 * 512 + 255) / 256, 256, 0, stream>>>(W1, Bpack1, 129, HC1, KT1, NT1);
    k_packB<<<(NT2 * KT2 * 512 + 255) / 256, 256, 0, stream>>>(W2, Bpack2, K2, C2, KT2, NT2);
    // 3) conv1 GEMM (WMMA) + attention scalars
    k_gemm<<<(MT * NT1 * 32) / 128, 128, 0, stream>>>(Apack1, Bpack1, xp1, NT1, KT1, HC1);
    k_attn1<<<(N_NODES * HEADS + 255) / 256, 256, 0, stream>>>(xp1, atts1, attd1, asrc1, adst1);
    // 4) conv1 edge softmax + scatter
    k_edge_max<<<(N_ETOT * HEADS + 255) / 256, 256, 0, stream>>>(ei, eattr, asrc1, adst1, coef1, m1, HEADS);
    k_edge_exp<<<(N_ETOT * HEADS + 255) / 256, 256, 0, stream>>>(ei, eattr, asrc1, adst1, coef1, m1, ex1, den1, HEADS);
    k_scatter1<<<N_ETOT, 256, 0, stream>>>(ei, xp1, ex1, den1, agg1);
    // 5) relu + bias, repack for conv2
    k_epi1<<<(MT * KT2 * 512 + 255) / 256, 256, 0, stream>>>(agg1, b1, Apack2);
    // 6) conv2 GEMM (WMMA) + attention scalars
    k_gemm<<<(MT * NT2 * 32) / 128, 128, 0, stream>>>(Apack2, Bpack2, xp2, NT2, KT2, C2);
    k_attn2<<<(N_NODES + 255) / 256, 256, 0, stream>>>(xp2, atts2, attd2, asrc2, adst2);
    // 7) conv2 edge softmax + scatter
    k_edge_max<<<(N_ETOT + 255) / 256, 256, 0, stream>>>(ei, eattr, asrc2, adst2, coef2, m2, 1);
    k_edge_exp<<<(N_ETOT + 255) / 256, 256, 0, stream>>>(ei, eattr, asrc2, adst2, coef2, m2, ex2, den2, 1);
    k_scatter2<<<(int)(((size_t)N_ETOT * C2 + 255) / 256), 256, 0, stream>>>(ei, xp2, ex2, den2, agg2);
    // 8) pooling
    k_pool<<<N_GRAPH, 64, 0, stream>>>(agg2, b2, pm, out);
}
